// MEGNet_79980880986464
// MI455X (gfx1250) — matrix operand order repaired and checked by reading, not compiled, not run on hardware
//
#include <hip/hip_runtime.h>
#include <math.h>

#define MG_NN 50000
#define MG_NE 800000

typedef __attribute__((ext_vector_type(16))) __bf16 v16bf;
typedef __attribute__((ext_vector_type(8)))  __bf16 v8bf;
typedef __attribute__((ext_vector_type(8)))  float  v8f;

union MgBFrag { v16bf v; v8bf h[2]; };
union MgFAcc  { v8f v; float f[8]; };
union MgStage { v8bf v[2]; __bf16 e[16]; };

__device__ __forceinline__ __bf16 mg_f2bf(float x) { return (__bf16)x; }

__device__ __forceinline__ float mg_softplus2(float x) {
  // softplus(x) - log(2), numerically stable
  return fmaxf(x, 0.0f) + __logf(1.0f + __expf(-fabsf(x))) - 0.69314718055994531f;
}

__device__ __forceinline__ v8f mg_wmma(v16bf a, v16bf b, v8f c) {
  return __builtin_amdgcn_wmma_f32_16x16x32_bf16(false, a, false, b, (short)0, c,
                                                 false, false);
}

// ---------------------------------------------------------------------------
// WMMA GEMM: Y[M,No] = act(A[M,Ni] @ W[No,Ni]^T + bias)
// A is either a plain f32 matrix X, or (nseg>0) a virtual concat of nseg
// 32-wide f32 segments with optional per-row gather (I0/I1) or broadcast.
// 256 threads = 8 waves; block tile = 128 rows; wave tile = 16 rows x No.
// ---------------------------------------------------------------------------
template <int NT>
__global__ __launch_bounds__(256) void mg_gemm(
    const float* __restrict__ X, const float* __restrict__ W,
    const float* __restrict__ Bz, float* __restrict__ Y,
    int M, int Ni, int No, int act, int nseg,
    const float* __restrict__ S0, const float* __restrict__ S1,
    const float* __restrict__ S2, const float* __restrict__ S3,
    const int* __restrict__ I0, const int* __restrict__ I1, int bcastMask)
{
  __shared__ __align__(16) __bf16 sW[64 * 128];
  __shared__ __align__(16) __bf16 sA[128 * 32];
  __shared__ float sBias[64];

  const int tid   = threadIdx.x;
  const int NiPad = (Ni + 31) & ~31;

  // Weights f32 -> bf16 LDS, zero-padded K
  for (int i = tid; i < No * NiPad; i += 256) {
    int n = i / NiPad, k = i - n * NiPad;
    sW[i] = mg_f2bf((k < Ni) ? W[(long)n * Ni + k] : 0.0f);
  }
  if (tid < No) sBias[tid] = Bz[tid];
  __syncthreads();

  const int wid   = tid >> 5;
  const int lane  = tid & 31;
  const int mBase = blockIdx.x * 128;

  MgFAcc acc[NT];
#pragma unroll
  for (int t = 0; t < NT; ++t)
#pragma unroll
    for (int r = 0; r < 8; ++r) acc[t].f[r] = 0.0f;

  const int sRow  = tid >> 1;         // 0..127
  const int sHalf = (tid & 1) * 16;   // 0 or 16
  const int gRow  = mBase + sRow;
  const bool rValid = gRow < M;

  for (int kk = 0; kk < NiPad; kk += 32) {
    // ---- stage A tile (128 x 32 bf16); each thread owns 16 contiguous cols
    MgStage st;
    if (nseg == 0) {
      const float* rowp = X + (long)gRow * Ni;
      const int c0 = kk + sHalf;
      if (rValid && (c0 + 15) < Ni) {
#pragma unroll
        for (int g = 0; g < 4; ++g) {
          float4 f = *(const float4*)(rowp + c0 + g * 4);   // 16B-aligned
          st.e[g * 4 + 0] = mg_f2bf(f.x);
          st.e[g * 4 + 1] = mg_f2bf(f.y);
          st.e[g * 4 + 2] = mg_f2bf(f.z);
          st.e[g * 4 + 3] = mg_f2bf(f.w);
        }
      } else {
#pragma unroll
        for (int j = 0; j < 16; ++j) {
          int c = c0 + j;
          st.e[j] = mg_f2bf((rValid && c < Ni) ? rowp[c] : 0.0f);
        }
      }
    } else {
      const int seg = kk >> 5;
      const float* Sp = (seg == 0) ? S0 : (seg == 1) ? S1 : (seg == 2) ? S2 : S3;
      const int*   Ip = (seg == 0) ? I0 : (seg == 1) ? I1 : nullptr;
      long r;
      if (bcastMask & (1 << seg)) r = 0;
      else if (Ip)                r = rValid ? (long)Ip[gRow] : 0;
      else                        r = gRow;
      if (rValid) {
        const float* rowp = Sp + r * 32 + sHalf;            // 16B-aligned
#pragma unroll
        for (int g = 0; g < 4; ++g) {
          float4 f = *(const float4*)(rowp + g * 4);
          st.e[g * 4 + 0] = mg_f2bf(f.x);
          st.e[g * 4 + 1] = mg_f2bf(f.y);
          st.e[g * 4 + 2] = mg_f2bf(f.z);
          st.e[g * 4 + 3] = mg_f2bf(f.w);
        }
      } else {
#pragma unroll
        for (int j = 0; j < 16; ++j) st.e[j] = mg_f2bf(0.0f);
      }
    }
    *(v8bf*)&sA[sRow * 32 + sHalf]     = st.v[0];
    *(v8bf*)&sA[sRow * 32 + sHalf + 8] = st.v[1];
    __syncthreads();

    // ---- A fragment: lane<16 -> K 0..7 & 16..23 ; lane>=16 -> K 8..15 & 24..31
    const int rl = (wid << 4) + (lane & 15);
    const int ko = (lane >> 4) << 3;
    MgBFrag af;
    af.h[0] = *(const v8bf*)&sA[rl * 32 + ko];
    af.h[1] = *(const v8bf*)&sA[rl * 32 + 16 + ko];

#pragma unroll
    for (int t = 0; t < NT; ++t) {
      const int n  = (t << 4) + (lane & 15);
      const int kb = (lane >> 4) << 3;    // lanes 0-15: K 0..7 & 16..23, lanes 16-31: K 8..15 & 24..31
      MgBFrag bf;
      bf.h[0] = *(const v8bf*)&sW[n * NiPad + kk + kb];
      bf.h[1] = *(const v8bf*)&sW[n * NiPad + kk + kb + 16];
      acc[t].v = mg_wmma(af.v, bf.v, acc[t].v);
    }
    __syncthreads();
  }

  // ---- epilogue: C layout VGPR r -> M=r (lanes 0-15) / M=8+r (lanes 16-31)
  const int rowOff  = mBase + (wid << 4) + ((lane >> 4) << 3);
  const int colLane = lane & 15;
#pragma unroll
  for (int t = 0; t < NT; ++t) {
    const int col = (t << 4) + colLane;
    const float bb = sBias[col];
#pragma unroll
    for (int r = 0; r < 8; ++r) {
      const int row = rowOff + r;
      if (row < M) {
        float y = acc[t].f[r] + bb;
        if (act) y = mg_softplus2(y);
        Y[(long)row * No + col] = y;
      }
    }
  }
}

// ---------------------------------------------------------------------------
// Elementwise / reduction helpers
// ---------------------------------------------------------------------------
__global__ void mg_zero(float* p, long n) {
  long i = (long)blockIdx.x * 256 + threadIdx.x;
  if (i < n) p[i] = 0.0f;
}
__global__ void mg_copy(float* d, const float* s, long n) {
  long i = (long)blockIdx.x * 256 + threadIdx.x;
  if (i < n) d[i] = s[i];
}
__global__ void mg_add(float* y, const float* a, const float* b, long n) {
  long i = (long)blockIdx.x * 256 + threadIdx.x;
  if (i < n) y[i] = a[i] + b[i];
}
__global__ void mg_init_i32(int* p, int v) { *p = v; }

__global__ void mg_embed(const float* __restrict__ emb, const int* __restrict__ tp,
                         float* __restrict__ out, int n) {
  long i = (long)blockIdx.x * 256 + threadIdx.x;
  if (i < (long)n * 16) {
    int node = (int)(i >> 4), d = (int)(i & 15);
    out[i] = emb[(long)tp[node] * 16 + d];
  }
}
__global__ void mg_degree(const int* __restrict__ dst, float* cnt, int e) {
  int i = blockIdx.x * 256 + threadIdx.x;
  if (i < e) atomicAdd(&cnt[dst[i]], 1.0f);
}
__global__ void mg_scatter_add(const float* __restrict__ m, const int* __restrict__ dst,
                               float* ve, long e32) {
  long i = (long)blockIdx.x * 256 + threadIdx.x;
  if (i < e32) {
    int e = (int)(i >> 5), d = (int)(i & 31);
    atomicAdd(&ve[(long)dst[e] * 32 + d], m[i]);
  }
}
__global__ void mg_div_cnt(float* ve, const float* __restrict__ cnt, long n32) {
  long i = (long)blockIdx.x * 256 + threadIdx.x;
  if (i < n32) ve[i] /= fmaxf(cnt[i >> 5], 1.0f);
}
__global__ __launch_bounds__(256) void mg_colsum(const float* __restrict__ x,
                                                 float* out32, long rows) {
  __shared__ float s[32];
  if (threadIdx.x < 32) s[threadIdx.x] = 0.0f;
  __syncthreads();
  long base   = (long)blockIdx.x * 256 + threadIdx.x;
  long stride = (long)gridDim.x * 256;            // multiple of 32 -> d is fixed
  long total  = rows * 32;
  float acc = 0.0f;
  for (long i = base; i < total; i += stride) acc += x[i];
  atomicAdd(&s[(int)(base & 31)], acc);
  __syncthreads();
  if (threadIdx.x < 32) atomicAdd(&out32[threadIdx.x], s[threadIdx.x]);
}
__global__ void mg_scale(float* p, int n, float s) {
  int i = threadIdx.x;
  if (i < n) p[i] *= s;
}

// ---------------------------------------------------------------------------
// Set2Set pieces
// ---------------------------------------------------------------------------
__global__ __launch_bounds__(128) void mg_lstm(
    const float* __restrict__ Wih, const float* __restrict__ Whh,
    const float* __restrict__ bih, const float* __restrict__ bhh,
    const float* __restrict__ qstar, float* h, float* c) {
  __shared__ float g[128], hs[32], qs[64];
  int t = threadIdx.x;
  if (t < 64) qs[t] = qstar[t];
  if (t < 32) hs[t] = h[t];
  __syncthreads();
  float a = bih[t] + bhh[t];
  for (int k = 0; k < 64; ++k) a += qs[k] * Wih[t * 64 + k];
  for (int k = 0; k < 32; ++k) a += hs[k] * Whh[t * 32 + k];
  g[t] = a;
  __syncthreads();
  if (t < 32) {
    float ig = 1.0f / (1.0f + __expf(-g[t]));
    float fg = 1.0f / (1.0f + __expf(-g[32 + t]));
    float gg = tanhf(g[64 + t]);
    float og = 1.0f / (1.0f + __expf(-g[96 + t]));
    float cn = fg * c[t] + ig * gg;
    c[t] = cn;
    h[t] = og * tanhf(cn);
  }
}
__global__ __launch_bounds__(256) void mg_dotmax(const float* __restrict__ feat,
                                                 const float* __restrict__ h,
                                                 float* dot, int* gmax, long rows) {
  __shared__ float hs[32];
  __shared__ int smax;
  if (threadIdx.x < 32) hs[threadIdx.x] = h[threadIdx.x];
  if (threadIdx.x == 0) smax = (int)0x80000000;
  __syncthreads();
  long i = (long)blockIdx.x * 256 + threadIdx.x;
  int enc = (int)0x80000000;
  if (i < rows) {
    float a = 0.0f;
    for (int d = 0; d < 32; ++d) a += feat[i * 32 + d] * hs[d];
    dot[i] = a;
    int e = __float_as_int(a);
    enc = (e >= 0) ? e : (e ^ 0x7FFFFFFF);        // order-preserving encoding
  }
  atomicMax(&smax, enc);
  __syncthreads();
  if (threadIdx.x == 0) atomicMax(gmax, smax);
}
__global__ __launch_bounds__(256) void mg_expsum(const float* __restrict__ feat,
                                                 const float* __restrict__ dot,
                                                 const int* __restrict__ gmax,
                                                 float* svec, long rows) {
  __shared__ float sv[33];
  if (threadIdx.x < 33) sv[threadIdx.x] = 0.0f;
  __syncthreads();
  int me = *gmax;
  float mx = __int_as_float((me >= 0) ? me : (me ^ 0x7FFFFFFF));
  long base   = (long)blockIdx.x * 256 + threadIdx.x;
  long stride = (long)gridDim.x * 256;
  long total  = rows * 32;
  int d = (int)(base & 31);
  float acc = 0.0f, accw = 0.0f;
  for (long i = base; i < total; i += stride) {
    long r = i >> 5;
    float w = __expf(dot[r] - mx);
    acc += w * feat[i];
    if (d == 0) accw += w;
  }
  atomicAdd(&sv[d], acc);
  if (d == 0) atomicAdd(&sv[32], accw);
  __syncthreads();
  if (threadIdx.x < 33) atomicAdd(&svec[threadIdx.x], sv[threadIdx.x]);
}
__global__ void mg_qstar(const float* h, const float* svec, float* qstar) {
  int t = threadIdx.x;
  if (t < 32) qstar[t] = h[t];
  else if (t < 64) qstar[t] = svec[t - 32] / svec[32];
}

// ---------------------------------------------------------------------------
// Tiny (M=1) MLP chain: out = chain(concat(s0,s1,s2)) [+ resid]
// ---------------------------------------------------------------------------
__global__ __launch_bounds__(64) void mg_small_mlp(
    const float* s0, int l0, const float* s1, int l1, const float* s2, int l2,
    const float* W0, const float* b0, int n0,
    const float* W1, const float* b1, int n1,
    const float* W2, const float* b2, int n2,
    int actLast, const float* resid, float* out)
{
  __shared__ float x[160], y[64];
  int t = threadIdx.x;
  for (int i = t; i < l0; i += 64) x[i] = s0[i];
  for (int i = t; i < l1; i += 64) x[l0 + i] = s1[i];
  for (int i = t; i < l2; i += 64) x[l0 + l1 + i] = s2[i];
  __syncthreads();
  int ni = l0 + l1 + l2;
  const float* Ws[3] = {W0, W1, W2};
  const float* bs[3] = {b0, b1, b2};
  int ns[3] = {n0, n1, n2};
  int nl = W2 ? 3 : (W1 ? 2 : 1);
  for (int L = 0; L < nl; ++L) {
    if (t < ns[L]) {
      float a = bs[L][t];
      for (int k = 0; k < ni; ++k) a += x[k] * Ws[L][t * ni + k];
      if (L < nl - 1 || actLast) a = mg_softplus2(a);
      y[t] = a;
    }
    __syncthreads();
    if (t < ns[L]) x[t] = y[t];
    ni = ns[L];
    __syncthreads();
  }
  if (t < ni) out[t] = x[t] + (resid ? resid[t] : 0.0f);
}

// ---------------------------------------------------------------------------
// Host side
// ---------------------------------------------------------------------------
static inline int mg_cdiv(long a, long b) { return (int)((a + b - 1) / b); }

static void mg_launch_gemm(hipStream_t st, const float* X, const float* W,
                           const float* B, float* Y, long M, int Ni, int No, int act,
                           int nseg = 0, const float* S0 = nullptr,
                           const float* S1 = nullptr, const float* S2 = nullptr,
                           const float* S3 = nullptr, const int* I0 = nullptr,
                           const int* I1 = nullptr, int bm = 0) {
  dim3 g(mg_cdiv(M, 128)), b(256);
  int Mi = (int)M;
  if (No == 16)
    mg_gemm<1><<<g, b, 0, st>>>(X, W, B, Y, Mi, Ni, No, act, nseg, S0, S1, S2, S3, I0, I1, bm);
  else if (No == 32)
    mg_gemm<2><<<g, b, 0, st>>>(X, W, B, Y, Mi, Ni, No, act, nseg, S0, S1, S2, S3, I0, I1, bm);
  else
    mg_gemm<4><<<g, b, 0, st>>>(X, W, B, Y, Mi, Ni, No, act, nseg, S0, S1, S2, S3, I0, I1, bm);
}

extern "C" void kernel_launch(void* const* d_in, const int* in_sizes, int n_in,
                              void* d_out, int out_size, void* d_ws, size_t ws_size,
                              hipStream_t stream) {
  (void)in_sizes; (void)n_in; (void)out_size; (void)ws_size;
  auto F = [&](int i) { return (const float*)d_in[i]; };
  auto Ii = [&](int i) { return (const int*)d_in[i]; };

  // jax pytree flatten order (dict keys sorted):
  // 0 edge_feat, 1 state_feat,
  // params: blocks[0].conv.{edge,node,state}=2..19,
  //         blocks[1].conv=20..37, edge_f=38..41, node_f=42..45, state_f=46..49,
  //         blocks[2].conv=50..67, edge_f=68..71, node_f=72..75, state_f=76..79,
  //         edge_enc=80..83, edge_s2s(Whh,Wih,bhh,bih)=84..87, node_emb=88,
  //         node_enc=89..92, node_s2s=93..96, out=97..102, state_enc=103..106,
  // 107 node_type, 108 src, 109 dst
  const float* edge_feat  = F(0);
  const float* state_feat = F(1);
  const int* node_type = Ii(107);
  const int* srcI = Ii(108);
  const int* dstI = Ii(109);

  const long NN = MG_NN, NE = MG_NE;
  float* ws = (float*)d_ws;
  long off = 0;
  auto A = [&](long n) { float* p = ws + off; off += n; return p; };
  float* u_cur  = A(32);
  float* u0     = A(32);
  float* uFb    = A(32);
  float* mean_m = A(32);
  float* mean_v = A(32);
  float* h_     = A(32);
  float* c_     = A(32);
  float* qstar  = A(64);
  float* qnode  = A(64);
  float* qedge  = A(64);
  float* svec   = A(64);            // 33 used
  int*   gmax   = (int*)A(16);
  float* cnt    = A(NN);
  float* v16b   = A(NN * 16);
  float* Va     = A(NN * 32);       // current v (== v0 until residual)
  float* Vf     = A(NN * 32);       // pre-MLP output
  float* Vve    = A(NN * 32);       // segment mean
  float* Vnew   = A(NN * 32);
  float* V64    = A(NN * 64);
  float* Ea     = A(NE * 32);       // current e (== e0 until residual)
  float* Ef     = A(NE * 32);
  float* Em     = A(NE * 32);
  float* E64    = A(NE * 64);
  float* dotb   = A(NE);

  const dim3 B256(256);

  // ---- encoders ----
  mg_embed<<<mg_cdiv(NN * 16, 256), B256, 0, stream>>>(F(88), node_type, v16b, (int)NN);
  mg_launch_gemm(stream, v16b, F(89), F(90), V64, NN, 16, 64, 1);
  mg_launch_gemm(stream, V64, F(91), F(92), Va, NN, 64, 32, 1);
  mg_launch_gemm(stream, edge_feat, F(80), F(81), E64, NE, 100, 64, 1);
  mg_launch_gemm(stream, E64, F(82), F(83), Ea, NE, 64, 32, 1);
  mg_small_mlp<<<1, 64, 0, stream>>>(state_feat, 2, nullptr, 0, nullptr, 0,
      F(103), F(104), 64, F(105), F(106), 32, nullptr, nullptr, 0, 1, nullptr, u_cur);
  mg_zero<<<mg_cdiv(NN, 256), B256, 0, stream>>>(cnt, NN);
  mg_degree<<<mg_cdiv(NE, 256), B256, 0, stream>>>(dstI, cnt, (int)NE);

  const int convB[3] = {2, 20, 50};
  const int fB[3]    = {0, 38, 68};
  for (int blk = 0; blk < 3; ++blk) {
    const int cb = convB[blk];
    mg_copy<<<1, 64, 0, stream>>>(u0, u_cur, 32);
    const float* ePre = Ea;
    const float* vPre = Va;
    const float* uPre = u_cur;
    if (blk > 0) {
      const int fb = fB[blk];
      mg_launch_gemm(stream, Ea, F(fb + 0), F(fb + 1), E64, NE, 32, 64, 1);
      mg_launch_gemm(stream, E64, F(fb + 2), F(fb + 3), Ef, NE, 64, 32, 1);
      ePre = Ef;
      mg_launch_gemm(stream, Va, F(fb + 4), F(fb + 5), V64, NN, 32, 64, 1);
      mg_launch_gemm(stream, V64, F(fb + 6), F(fb + 7), Vf, NN, 64, 32, 1);
      vPre = Vf;
      mg_small_mlp<<<1, 64, 0, stream>>>(u_cur, 32, nullptr, 0, nullptr, 0,
          F(fb + 8), F(fb + 9), 64, F(fb + 10), F(fb + 11), 32,
          nullptr, nullptr, 0, 1, nullptr, uFb);
      uPre = uFb;
    }
    // edge messages: concat(v[src], v[dst], e, u) -> 64 -> 64 -> 32
    mg_launch_gemm(stream, nullptr, F(cb + 0), F(cb + 1), E64, NE, 128, 64, 1,
                   4, vPre, vPre, ePre, uPre, srcI, dstI, 0x8);
    mg_launch_gemm(stream, E64, F(cb + 2), F(cb + 3), E64, NE, 64, 64, 1);
    mg_launch_gemm(stream, E64, F(cb + 4), F(cb + 5), Em, NE, 64, 32, 1);
    // ve = mean of incoming messages
    mg_zero<<<mg_cdiv(NN * 32, 256), B256, 0, stream>>>(Vve, NN * 32);
    mg_scatter_add<<<mg_cdiv(NE * 32, 256), B256, 0, stream>>>(Em, dstI, Vve, NE * 32);
    mg_div_cnt<<<mg_cdiv(NN * 32, 256), B256, 0, stream>>>(Vve, cnt, NN * 32);
    // mean over edges of m
    mg_zero<<<1, 64, 0, stream>>>(mean_m, 32);
    mg_colsum<<<1024, B256, 0, stream>>>(Em, mean_m, NE);
    mg_scale<<<1, 64, 0, stream>>>(mean_m, 32, 1.0f / (float)NE);
    // node update: concat(v, ve, u) -> 64 -> 64 -> 32
    mg_launch_gemm(stream, nullptr, F(cb + 6), F(cb + 7), V64, NN, 96, 64, 1,
                   3, vPre, Vve, uPre, nullptr, nullptr, nullptr, 0x4);
    mg_launch_gemm(stream, V64, F(cb + 8), F(cb + 9), V64, NN, 64, 64, 1);
    mg_launch_gemm(stream, V64, F(cb + 10), F(cb + 11), Vnew, NN, 64, 32, 1);
    // mean over nodes of v_new
    mg_zero<<<1, 64, 0, stream>>>(mean_v, 32);
    mg_colsum<<<512, B256, 0, stream>>>(Vnew, mean_v, NN);
    mg_scale<<<1, 64, 0, stream>>>(mean_v, 32, 1.0f / (float)NN);
    // state update (+residual u0)
    mg_small_mlp<<<1, 64, 0, stream>>>(uPre, 32, mean_m, 32, mean_v, 32,
        F(cb + 12), F(cb + 13), 64, F(cb + 14), F(cb + 15), 64,
        F(cb + 16), F(cb + 17), 32, 1, u0, u_cur);
    // residuals e += m, v += v_new (in place, elementwise)
    mg_add<<<mg_cdiv(NE * 32, 256), B256, 0, stream>>>(Ea, Em, Ea, NE * 32);
    mg_add<<<mg_cdiv(NN * 32, 256), B256, 0, stream>>>(Va, Vnew, Va, NN * 32);
  }

  // ---- Set2Set over nodes then edges ----
  for (int which = 0; which < 2; ++which) {
    const float* feat = which ? Ea : Va;
    long rows = which ? NE : NN;
    const float* Wih = which ? F(85) : F(94);
    const float* Whh = which ? F(84) : F(93);
    const float* bih = which ? F(87) : F(96);
    const float* bhh = which ? F(86) : F(95);
    float* qout = which ? qedge : qnode;
    mg_zero<<<1, 64, 0, stream>>>(qstar, 64);
    mg_zero<<<1, 64, 0, stream>>>(h_, 32);
    mg_zero<<<1, 64, 0, stream>>>(c_, 32);
    for (int it = 0; it < 2; ++it) {
      mg_lstm<<<1, 128, 0, stream>>>(Wih, Whh, bih, bhh, qstar, h_, c_);
      mg_init_i32<<<1, 1, 0, stream>>>(gmax, (int)0x80000000);
      mg_zero<<<1, 64, 0, stream>>>(svec, 33);
      mg_dotmax<<<mg_cdiv(rows, 256), B256, 0, stream>>>(feat, h_, dotb, gmax, rows);
      mg_expsum<<<1024, B256, 0, stream>>>(feat, dotb, gmax, svec, rows);
      mg_qstar<<<1, 64, 0, stream>>>(h_, svec, qstar);
    }
    mg_copy<<<1, 64, 0, stream>>>(qout, qstar, 64);
  }

  // ---- output head: concat(qnode, qedge, u) -> 32 -> 16 -> 1 (no act last) ----
  mg_small_mlp<<<1, 64, 0, stream>>>(qnode, 64, qedge, 64, u_cur, 32,
      F(97), F(98), 32, F(99), F(100), 16, F(101), F(102), 1,
      0, nullptr, (float*)d_out);
}
